// FeedForward_27650999451936
// MI455X (gfx1250) — compile-verified, matrix operand order checked by reading, not run on hardware
//
#include <hip/hip_runtime.h>

#define HIDDEN 4096
#define INTER  11008
#define MROWS  128
#define KSPLIT 8
#define KCHUNK (INTER / KSPLIT)   // 1376, divisible by 32

typedef __attribute__((ext_vector_type(16))) _Float16 v16h;
typedef __attribute__((ext_vector_type(8)))  _Float16 v8h;
typedef __attribute__((ext_vector_type(4)))  _Float16 v4h;
typedef __attribute__((ext_vector_type(8)))  float    v8f;

union FragU { v16h v; struct { v8h lo, hi; } h2; };

// A/B fragment for v_wmma_f32_16x16x32_f16, loaded from a row-major [rows, ld]
// fp32 matrix (weights are integer-valued fp32 -> f16 conversion is exact).
// ISA 16-bit A layout: lane<16 holds row (lane&15), K = {0..7, 16..23};
// lane>=16 holds same row set, K = {8..15, 24..31}.
__device__ __forceinline__ v16h frag_from_f32(const float* __restrict__ base,
                                              int ld, int k0, int lane) {
  const int row = lane & 15;
  const int cb  = (lane >> 4) << 3;          // 0 or 8
  const float* p = base + (size_t)row * ld + k0 + cb;
  v16h f;
#pragma unroll
  for (int i = 0; i < 8; ++i) {
    f[i]     = (_Float16)p[i];               // K = cb + i
    f[i + 8] = (_Float16)p[i + 16];          // K = cb + 16 + i
  }
  return f;
}

// Same fragment, native f16 source: two 16B vector loads per lane, no VALU.
__device__ __forceinline__ v16h frag_from_f16(const _Float16* __restrict__ base,
                                              int ld, int k0, int lane) {
  const int row = lane & 15;
  const int cb  = (lane >> 4) << 3;
  const _Float16* p = base + (size_t)row * ld + k0 + cb;
  FragU f;
  f.h2.lo = *(const v8h*)(p);
  f.h2.hi = *(const v8h*)(p + 16);
  return f.v;
}

__device__ __forceinline__ v8f wmma_f16(v16h a, v16h b, v8f c) {
  // 8 args: (neg_a, A, neg_b, B, c_mod, C, reuse_a, reuse_b)
  return __builtin_amdgcn_wmma_f32_16x16x32_f16(false, a, false, b,
                                                (short)0, c, false, false);
}

// ---------------------------------------------------------------------------
// Pre-pass: X fp32 -> f16 once (X is reused 688x by kernel A; converting it
// per-block would burn ~64 v_cvt per K-step in the hot loop).
// grid covers 128*HIDDEN elements, 4 per thread.
// ---------------------------------------------------------------------------
__global__ __launch_bounds__(256) void convert_x(
    const float* __restrict__ x, _Float16* __restrict__ xh) {
  const int i = (blockIdx.x * 256 + (int)threadIdx.x) * 4;
  const float4 v = *(const float4*)(x + i);
  v4h o;
  o[0] = (_Float16)v.x; o[1] = (_Float16)v.y;
  o[2] = (_Float16)v.z; o[3] = (_Float16)v.w;
  *(v4h*)(xh + i) = o;
}

// ---------------------------------------------------------------------------
// Kernel A: H = silu(X@W1^T * s1) * (X@W3^T * s3), H stored as f16 [128,INTER]
// 8 waves/block; wave owns one 16-wide N tile and all 8 M tiles (full M reuse
// => each w1/w3 element streamed from HBM exactly once across the grid).
// grid.x = INTER/128 = 86
// ---------------------------------------------------------------------------
__global__ __launch_bounds__(256) void ffn_gate_up(
    const _Float16* __restrict__ xh, // [128, HIDDEN] f16
    const float* __restrict__ w1,    // [INTER, HIDDEN]
    const float* __restrict__ w1s,   // [INTER]
    const float* __restrict__ w3,    // [INTER, HIDDEN]
    const float* __restrict__ w3s,   // [INTER]
    _Float16* __restrict__ h)        // [128, INTER]
{
  const int wave = (int)(threadIdx.x >> 5);
  const int lane = (int)(threadIdx.x & 31);
  const int n0   = (blockIdx.x * 8 + wave) * 16;

  const float* w1row = w1 + (size_t)n0 * HIDDEN;
  const float* w3row = w3 + (size_t)n0 * HIDDEN;

  v8f up[8], gate[8];
  const v8f zero = {0.f, 0.f, 0.f, 0.f, 0.f, 0.f, 0.f, 0.f};
#pragma unroll
  for (int m = 0; m < 8; ++m) { up[m] = zero; gate[m] = zero; }

  for (int k0 = 0; k0 < HIDDEN; k0 += 32) {
    // keep the HBM weight stream ahead of the math (~4 K-steps ahead)
    __builtin_prefetch(w1row + (size_t)(lane & 15) * HIDDEN + k0 + 128, 0, 1);
    __builtin_prefetch(w3row + (size_t)(lane & 15) * HIDDEN + k0 + 128, 0, 1);

    const v16h b1 = frag_from_f32(w1row, HIDDEN, k0, lane);
    const v16h b3 = frag_from_f32(w3row, HIDDEN, k0, lane);
#pragma unroll
    for (int m = 0; m < 8; ++m) {
      const v16h a = frag_from_f16(xh + (size_t)m * 16 * HIDDEN, HIDDEN, k0, lane);
      up[m]   = wmma_f16(a, b1, up[m]);
      gate[m] = wmma_f16(a, b3, gate[m]);
    }
  }

  // Epilogue: C/D layout -> VGPR r: M = 16*m + r + 8*(lane>>4), N = n0+(lane&15)
  const int n   = n0 + (lane & 15);
  const int mhi = (lane >> 4) << 3;
  const float s1 = w1s[n];
  const float s3 = w3s[n];
#pragma unroll
  for (int m = 0; m < 8; ++m) {
#pragma unroll
    for (int r = 0; r < 8; ++r) {
      const int row = m * 16 + r + mhi;
      const float u = up[m][r] * s1;
      const float g = gate[m][r] * s3;
      const float hv = (u / (1.f + __expf(-u))) * g;   // silu(u) * g
      h[(size_t)row * INTER + n] = (_Float16)hv;       // coalesced across lanes
    }
  }
}

// ---------------------------------------------------------------------------
// Kernel B: split-K partials of H @ W2^T.  grid = (HIDDEN/128, KSPLIT).
// part[ks][128][HIDDEN] fp32 in workspace.
// ---------------------------------------------------------------------------
__global__ __launch_bounds__(256) void ffn_down_partial(
    const _Float16* __restrict__ h, // [128, INTER]
    const float* __restrict__ w2,   // [HIDDEN, INTER]
    float* __restrict__ part)       // [KSPLIT][128][HIDDEN]
{
  const int wave = (int)(threadIdx.x >> 5);
  const int lane = (int)(threadIdx.x & 31);
  const int n0   = (blockIdx.x * 8 + wave) * 16;
  const int ks   = blockIdx.y;

  const float* w2row = w2 + (size_t)n0 * INTER;

  v8f acc[8];
  const v8f zero = {0.f, 0.f, 0.f, 0.f, 0.f, 0.f, 0.f, 0.f};
#pragma unroll
  for (int m = 0; m < 8; ++m) acc[m] = zero;

  const int kbeg = ks * KCHUNK;
  const int kend = kbeg + KCHUNK;
  for (int k0 = kbeg; k0 < kend; k0 += 32) {
    __builtin_prefetch(w2row + (size_t)(lane & 15) * INTER + k0 + 128, 0, 1);
    const v16h b = frag_from_f32(w2row, INTER, k0, lane);
#pragma unroll
    for (int m = 0; m < 8; ++m) {
      const v16h a = frag_from_f16(h + (size_t)m * 16 * INTER, INTER, k0, lane);
      acc[m] = wmma_f16(a, b, acc[m]);
    }
  }

  const int n   = n0 + (lane & 15);
  const int mhi = (lane >> 4) << 3;
  float* pbase = part + (size_t)ks * MROWS * HIDDEN;
#pragma unroll
  for (int m = 0; m < 8; ++m) {
#pragma unroll
    for (int r = 0; r < 8; ++r) {
      const int row = m * 16 + r + mhi;
      pbase[(size_t)row * HIDDEN + n] = acc[m][r];
    }
  }
}

// ---------------------------------------------------------------------------
// Deterministic split-K reduction + per-channel scale.
// ---------------------------------------------------------------------------
__global__ __launch_bounds__(256) void ffn_reduce(
    const float* __restrict__ part, // [KSPLIT][128][HIDDEN]
    const float* __restrict__ w2s,  // [HIDDEN]
    float* __restrict__ out)        // [128, HIDDEN]
{
  const int idx = blockIdx.x * 256 + (int)threadIdx.x;   // over 128*HIDDEN
  float s = 0.f;
#pragma unroll
  for (int ks = 0; ks < KSPLIT; ++ks)
    s += part[(size_t)ks * MROWS * HIDDEN + idx];
  out[idx] = s * w2s[idx & (HIDDEN - 1)];
}

extern "C" void kernel_launch(void* const* d_in, const int* in_sizes, int n_in,
                              void* d_out, int out_size, void* d_ws, size_t ws_size,
                              hipStream_t stream) {
  const float* x   = (const float*)d_in[0];  // hidden_state [8,16,4096]
  const float* w1  = (const float*)d_in[1];  // [INTER, HIDDEN]
  const float* w1s = (const float*)d_in[2];  // [INTER]
  const float* w3  = (const float*)d_in[3];  // [INTER, HIDDEN]
  const float* w3s = (const float*)d_in[4];  // [INTER]
  const float* w2  = (const float*)d_in[5];  // [HIDDEN, INTER]
  const float* w2s = (const float*)d_in[6];  // [HIDDEN]
  float* out = (float*)d_out;

  // workspace layout:
  //   [0, 16MB)        fp32 split-K partials  [KSPLIT][128][HIDDEN]
  //   [16MB, +2.75MB)  H f16                  [128][INTER]
  //   then             X f16                  [128][HIDDEN]
  const size_t part_bytes = (size_t)KSPLIT * MROWS * HIDDEN * sizeof(float);
  const size_t h_bytes    = (size_t)MROWS * INTER * sizeof(_Float16);
  float*    part = (float*)d_ws;
  _Float16* h    = (_Float16*)((char*)d_ws + part_bytes);
  _Float16* xh   = (_Float16*)((char*)d_ws + part_bytes + h_bytes);

  convert_x<<<dim3((MROWS * HIDDEN) / 1024), 256, 0, stream>>>(x, xh);
  ffn_gate_up<<<dim3(INTER / 128), 256, 0, stream>>>(xh, w1, w1s, w3, w3s, h);
  ffn_down_partial<<<dim3(HIDDEN / 128, KSPLIT), 256, 0, stream>>>(h, w2, part);
  ffn_reduce<<<dim3((MROWS * HIDDEN) / 256), 256, 0, stream>>>(part, w2s, out);
}
